// SpecAdaConv2d_7103875907597
// MI455X (gfx1250) — matrix operand order repaired and checked by reading, not compiled, not run on hardware
//
#include <hip/hip_runtime.h>
#include <hip/hip_bf16.h>
#include <stdint.h>

typedef __attribute__((ext_vector_type(16))) __bf16 v16bf;
typedef __attribute__((ext_vector_type(8)))  float  v8f;
typedef __attribute__((ext_vector_type(4)))  unsigned int v4u;

#define CIN   64
#define COUT  64
#define HWDIM 256
#define BATCH 8
#define LDS_S 72   // padded ci-stride (elements): 144B/column -> 36n mod 64 bank permutation

__device__ __forceinline__ unsigned short f2bf(float f) {
    unsigned int u = __float_as_uint(f);
    u += 0x7FFFu + ((u >> 16) & 1u);          // round-to-nearest-even
    return (unsigned short)(u >> 16);
}

// xs[b][h][w][ci] = bf16(alpha[b][ci] * x[b][ci][h][w])   (ci innermost for b128 gathers)
__global__ __launch_bounds__(256) void prescale_kernel(
    const float* __restrict__ x, const float* __restrict__ alpha,
    unsigned short* __restrict__ xs) {
    const int b  = blockIdx.z;
    const int h  = blockIdx.y;
    const int w  = blockIdx.x * 32 + (threadIdx.x & 31);
    const int ci0 = (threadIdx.x >> 5) * 8;

    float a[8];
#pragma unroll
    for (int j = 0; j < 8; ++j) a[j] = alpha[b * CIN + ci0 + j];

    unsigned int p[4];
#pragma unroll
    for (int k = 0; k < 4; ++k) {
        const int ci = ci0 + 2 * k;
        const float v0 = x[(((size_t)b * CIN + ci    ) * HWDIM + h) * HWDIM + w] * a[2 * k];
        const float v1 = x[(((size_t)b * CIN + ci + 1) * HWDIM + h) * HWDIM + w] * a[2 * k + 1];
        p[k] = (unsigned)f2bf(v0) | ((unsigned)f2bf(v1) << 16);
    }
    v4u val = { p[0], p[1], p[2], p[3] };
    *(v4u*)(xs + ((((size_t)b * HWDIM + h) * HWDIM) + w) * CIN + ci0) = val;
}

// wrs[pos][cout][ci] = bf16(weight[cout][ci][kh][kw]),  pos = kh*3+kw
__global__ __launch_bounds__(256) void wprep_kernel(
    const float* __restrict__ weight, unsigned short* __restrict__ wrs) {
    const int idx = blockIdx.x * 256 + threadIdx.x;
    if (idx >= 9 * COUT * CIN) return;
    const int pos  = idx / (COUT * CIN);
    const int rem  = idx - pos * (COUT * CIN);
    const int cout = rem >> 6;
    const int ci   = rem & 63;
    wrs[idx] = f2bf(weight[(cout * CIN + ci) * 9 + pos]);
}

// Implicit-GEMM conv. Block = (b, h, 128-wide w strip) -> 64 couts x 128 pixels.
// 8 waves = 4 cout-tiles x 2 pixel-halves; each wave: 4 accumulators sharing one A fragment.
__global__ __launch_bounds__(256) void conv_wmma_kernel(
    const unsigned short* __restrict__ xs, const unsigned short* __restrict__ wrs,
    const float* __restrict__ bias, float* __restrict__ out) {
    __shared__ unsigned short tile[3 * 130 * LDS_S];  // [row 0..2][col 0..129][ci 0..63] padded

    const int b   = blockIdx.z;
    const int h   = blockIdx.y;
    const int w0  = blockIdx.x * 128;
    const int tid = threadIdx.x;

    // ---- stage input halo tile (3 x 130 x 64ci bf16) into LDS, zero-padded borders ----
    for (int c = tid; c < 3 * 130 * 8; c += 256) {    // 390 columns * 8 chunks of 8 ci
        const int rowcol = c >> 3;
        const int ci0    = (c & 7) * 8;
        const int row    = rowcol / 130;
        const int col    = rowcol - row * 130;
        const int gh     = h + row - 1;
        const int gw     = w0 + col - 1;
        v4u val = { 0u, 0u, 0u, 0u };
        if ((unsigned)gh < (unsigned)HWDIM && (unsigned)gw < (unsigned)HWDIM) {
            val = *(const v4u*)(xs + ((((size_t)b * HWDIM + gh) * HWDIM) + gw) * CIN + ci0);
        }
        *(v4u*)(tile + rowcol * LDS_S + ci0) = val;
    }
    __syncthreads();

    // ---- per-wave: 16 couts x 64 pixels (four 16x16 WMMA tiles), K = 576 ----
    const int lane  = tid & 31;
    const int wave  = tid >> 5;
    const int mtile = wave & 3;            // cout tile 0..3
    const int nhalf = wave >> 2;           // pixel half 0..1 -> 64-px span
    const int n     = lane & 15;
    const int half  = lane >> 4;

    // A (16x32 bf16): lane m = lane&15; elem e holds K = half*16 + e
    const int aBase = (mtile * 16 + n) * CIN + half * 16;
    // B (32x16 bf16): lane n; elem e holds K = half*16 + e -> 16 contiguous ci
    const int bCol0 = nhalf * 64 + n;      // tiles at col +0, +16, +32, +48
    const int bCi   = half * 16;

    v8f acc[4] = {};
#pragma unroll
    for (int pos = 0; pos < 9; ++pos) {
        const int kh = pos / 3, kw = pos - 3 * kh;
        const unsigned short* wp  = wrs + pos * (COUT * CIN) + aBase;
        const unsigned short* bp0 = tile + (kh * 130 + bCol0 + kw) * LDS_S + bCi;
#pragma unroll
        for (int kg = 0; kg < 2; ++kg) {
            union { v16bf v; v4u q[2]; } A;
            A.q[0] = *(const v4u*)(wp + kg * 32);        // K = kg*32 + half*16 + 0..7
            A.q[1] = *(const v4u*)(wp + kg * 32 + 8);    // K = kg*32 + half*16 + 8..15
#pragma unroll
            for (int t = 0; t < 4; ++t) {
                union { v16bf v; v4u q[2]; } B;
                B.q[0] = *(const v4u*)(bp0 + t * 16 * LDS_S + kg * 32);      // K = kg*32+half*16+0..7
                B.q[1] = *(const v4u*)(bp0 + t * 16 * LDS_S + kg * 32 + 8);  // K = kg*32+half*16+8..15
                acc[t] = __builtin_amdgcn_wmma_f32_16x16x32_bf16(
                    false, A.v, false, B.v, (short)0, acc[t], false, false);
            }
        }
    }

    // ---- C/D layout: vgpr r, lane -> (m = r + half*8, n); add bias, store all 4 tiles ----
    float bs[8];
#pragma unroll
    for (int r = 0; r < 8; ++r) bs[r] = bias[mtile * 16 + half * 8 + r];

#pragma unroll
    for (int r = 0; r < 8; ++r) {
        const int m = mtile * 16 + half * 8 + r;
        const size_t rowbase = (((size_t)b * COUT + m) * HWDIM + h) * HWDIM + w0 + nhalf * 64 + n;
#pragma unroll
        for (int t = 0; t < 4; ++t) {
            out[rowbase + t * 16] = acc[t][r] + bs[r];
        }
    }
}

extern "C" void kernel_launch(void* const* d_in, const int* in_sizes, int n_in,
                              void* d_out, int out_size, void* d_ws, size_t ws_size,
                              hipStream_t stream) {
    const float* x      = (const float*)d_in[0];
    const float* alpha  = (const float*)d_in[1];
    const float* weight = (const float*)d_in[2];
    const float* bias   = (const float*)d_in[3];
    float* out = (float*)d_out;

    unsigned short* xs  = (unsigned short*)d_ws;                       // 64 MB bf16 prescaled x
    unsigned short* wrs = xs + (size_t)BATCH * HWDIM * HWDIM * CIN;    // 72 KB bf16 repacked w

    prescale_kernel<<<dim3(HWDIM / 32, HWDIM, BATCH), 256, 0, stream>>>(x, alpha, xs);
    wprep_kernel<<<dim3((9 * COUT * CIN + 255) / 256), 256, 0, stream>>>(weight, wrs);
    conv_wmma_kernel<<<dim3(HWDIM / 128, HWDIM, BATCH), 256, 0, stream>>>(xs, wrs, bias, out);
}
